// SummGCN_90409061580816
// MI455X (gfx1250) — compile-verified, matrix operand order checked by reading, not run on hardware
//
#include <hip/hip_runtime.h>
#include <math.h>

// ---------- problem constants (match reference) ----------
#define IN_DIM  500
#define HID_DIM 128
#define OUT_DIM 40

#define KSTEPS1 16            // ceil(500/32)
#define NTILES1 (HID_DIM/16)  // 8
#define KSTEPS2 (HID_DIM/32)  // 4
#define NTILES2 3             // ceil(40/16)

// ---------- CDNA5 WMMA types ----------
typedef __attribute__((ext_vector_type(16))) __bf16 v16bf;
typedef __attribute__((ext_vector_type(8)))  float  v8f;

// ============================================================
// helpers
// ============================================================
__device__ __forceinline__ v16bf make_frag(float4 q0, float4 q1, float4 q2, float4 q3) {
    v16bf a;
    a[0]  = (__bf16)q0.x; a[1]  = (__bf16)q0.y; a[2]  = (__bf16)q0.z; a[3]  = (__bf16)q0.w;
    a[4]  = (__bf16)q1.x; a[5]  = (__bf16)q1.y; a[6]  = (__bf16)q1.z; a[7]  = (__bf16)q1.w;
    a[8]  = (__bf16)q2.x; a[9]  = (__bf16)q2.y; a[10] = (__bf16)q2.z; a[11] = (__bf16)q2.w;
    a[12] = (__bf16)q3.x; a[13] = (__bf16)q3.y; a[14] = (__bf16)q3.z; a[15] = (__bf16)q3.w;
    return a;
}
__device__ __forceinline__ float4 relu4(float4 q) {
    q.x = fmaxf(q.x, 0.f); q.y = fmaxf(q.y, 0.f);
    q.z = fmaxf(q.z, 0.f); q.w = fmaxf(q.w, 0.f);
    return q;
}

// ============================================================
// zero-fill
// ============================================================
__global__ void zero_f32(float* __restrict__ p, long n) {
    long i = (long)blockIdx.x * blockDim.x + threadIdx.x;
    long stride = (long)gridDim.x * blockDim.x;
    for (; i < n; i += stride) p[i] = 0.0f;
}

// ============================================================
// Pre-pack B fragments: per (nTile, kstep, lane) store the exact
// v16bf WMMA B-fragment contiguously (zero-padded for K>=Krows or
// col>=Ncols) so the GEMM hot loop does a single 32B vector load.
// Layout index: ((nT*kSteps + ks)*32 + lane)*16 bf16 elements.
// ============================================================
__global__ void packB(const float* __restrict__ W, __bf16* __restrict__ Bp,
                      int Krows, int Ncols, int nTiles, int kSteps) {
    int idx = blockIdx.x * blockDim.x + threadIdx.x;
    int total = nTiles * kSteps * 32;
    if (idx >= total) return;
    int lane = idx & 31;
    int ks   = (idx >> 5) % kSteps;
    int nT   = (idx >> 5) / kSteps;
    int half = lane >> 4;
    int col  = nT * 16 + (lane & 15);
    int k0   = ks * 32;
    __bf16* dst = Bp + (size_t)idx * 16;
#pragma unroll
    for (int e = 0; e < 16; ++e) {
        int kk = k0 + ((e < 8) ? (8 * half + e) : (16 + 8 * half + (e - 8)));
        float f = (kk < Krows && col < Ncols) ? W[(size_t)kk * Ncols + col] : 0.0f;
        dst[e] = (__bf16)f;
    }
}

// ============================================================
// GEMM1: H[N,128] = X[N,500] @ W1[500,128]   (bf16 WMMA, f32 acc)
// block = 128 (4 waves); wave w -> nTile = blockIdx.y*4 + w; grid (N/16, 2)
// A: 4 unconditional b128 loads/step (lane's K runs are contiguous).
// B: one 32B packed-fragment load/step.
// ============================================================
__global__ void gemm1_wmma(const float* __restrict__ X,
                           const __bf16* __restrict__ Bp,
                           float* __restrict__ H, int nRows) {
    const int lane  = threadIdx.x & 31;
    const int wave  = threadIdx.x >> 5;
    const int mTile = blockIdx.x;
    const int nTile = blockIdx.y * 4 + wave;      // 0..7
    const int half  = lane >> 4;

    int row = mTile * 16 + (lane & 15);
    const bool rowStore = (row < nRows);
    if (row >= nRows) row = nRows - 1;            // clamp: stores masked below
    const float* __restrict__ xr = X + (size_t)row * IN_DIM;
    const __bf16* __restrict__ bp =
        Bp + ((size_t)(nTile * KSTEPS1) * 32 + lane) * 16;

    v8f acc = {};
    int ks = 0;
    for (int k0 = 0; k0 + 32 <= IN_DIM; k0 += 32, ++ks) {   // 15 full steps
        const int ka = k0 + 8 * half;
        float4 q0 = *(const float4*)(xr + ka);
        float4 q1 = *(const float4*)(xr + ka + 4);
        float4 q2 = *(const float4*)(xr + ka + 16);
        float4 q3 = *(const float4*)(xr + ka + 20);
        __builtin_prefetch(xr + ka + 32, 0, 1);             // global_prefetch_b8
        v16bf a = make_frag(q0, q1, q2, q3);
        v16bf b = *(const v16bf*)(bp + (size_t)ks * 32 * 16);
        acc = __builtin_amdgcn_wmma_f32_16x16x32_bf16(false, a, false, b,
                                                      (short)0, acc, false, false);
    }
    {   // K tail: 480..499, float4-granularity guards (2 small branches)
        const int k0 = ks * 32;                             // 480
        const int ka = k0 + 8 * half;
        const float4 z = make_float4(0.f, 0.f, 0.f, 0.f);
        float4 q0 = (ka      + 3 < IN_DIM) ? *(const float4*)(xr + ka)      : z;
        float4 q1 = (ka + 4  + 3 < IN_DIM) ? *(const float4*)(xr + ka + 4)  : z;
        float4 q2 = (ka + 16 + 3 < IN_DIM) ? *(const float4*)(xr + ka + 16) : z;
        float4 q3 = (ka + 20 + 3 < IN_DIM) ? *(const float4*)(xr + ka + 20) : z;
        v16bf a = make_frag(q0, q1, q2, q3);
        v16bf b = *(const v16bf*)(bp + (size_t)ks * 32 * 16);   // pre-zeroed pad
        acc = __builtin_amdgcn_wmma_f32_16x16x32_bf16(false, a, false, b,
                                                      (short)0, acc, false, false);
    }
    // D 16x16 f32: lane -> col (lane&15), VGPR v -> row v + 8*half
#pragma unroll
    for (int v = 0; v < 8; ++v) {
        int m = mTile * 16 + v + 8 * half;
        int n = nTile * 16 + (lane & 15);
        if (m < nRows) H[(size_t)m * HID_DIM + n] = acc[v];
    }
    (void)rowStore;
}

// ============================================================
// GEMM2: H2[N,40] = relu(H1[N,128]) @ W2[128,40]
// block = 96 (3 waves); wave w -> nTile w; grid (N/16). K=128 exact.
// ============================================================
__global__ void gemm2_wmma(const float* __restrict__ H1,
                           const __bf16* __restrict__ Bp,
                           float* __restrict__ H2, int nRows) {
    const int lane  = threadIdx.x & 31;
    const int wave  = threadIdx.x >> 5;            // 0..2
    const int mTile = blockIdx.x;
    const int nTile = wave;
    const int half  = lane >> 4;

    int row = mTile * 16 + (lane & 15);
    if (row >= nRows) row = nRows - 1;             // clamp: stores masked below
    const float* __restrict__ hr = H1 + (size_t)row * HID_DIM;
    const __bf16* __restrict__ bp =
        Bp + ((size_t)(nTile * KSTEPS2) * 32 + lane) * 16;

    v8f acc = {};
#pragma unroll
    for (int ks = 0; ks < KSTEPS2; ++ks) {
        const int ka = ks * 32 + 8 * half;
        float4 q0 = relu4(*(const float4*)(hr + ka));
        float4 q1 = relu4(*(const float4*)(hr + ka + 4));
        float4 q2 = relu4(*(const float4*)(hr + ka + 16));
        float4 q3 = relu4(*(const float4*)(hr + ka + 20));
        v16bf a = make_frag(q0, q1, q2, q3);
        v16bf b = *(const v16bf*)(bp + (size_t)ks * 32 * 16);  // cols>=40 pre-zeroed
        acc = __builtin_amdgcn_wmma_f32_16x16x32_bf16(false, a, false, b,
                                                      (short)0, acc, false, false);
    }
#pragma unroll
    for (int v = 0; v < 8; ++v) {
        int m = mTile * 16 + v + 8 * half;
        int n = nTile * 16 + (lane & 15);
        if (m < nRows && n < OUT_DIM) H2[(size_t)m * OUT_DIM + n] = acc[v];
    }
}

// ============================================================
// COO SpMM, rows sorted: Out[r,:] += sum_e vals[e] * H[cols[e],:]
// Thread t owns feature t; run-length accumulate per row in registers,
// atomicAdd (L2-resident) only at row boundaries.
// ============================================================
template <int D>
__global__ void spmm_coo(const int* __restrict__ rows,
                         const int* __restrict__ cols,
                         const float* __restrict__ vals,
                         const float* __restrict__ H,
                         float* __restrict__ Out,
                         int nE, int chunk) {
    const int t = threadIdx.x;
    if (t >= D) return;
    long e0 = (long)blockIdx.x * chunk;
    long e1 = e0 + chunk; if (e1 > nE) e1 = nE;

    float acc = 0.0f;
    int cur = -1;
    for (long e = e0; e < e1; ++e) {
        int rr = rows[e];
        if (rr != cur) {
            if (cur >= 0) atomicAdd(&Out[(long)cur * D + t], acc);
            cur = rr;
            acc = 0.0f;
        }
        acc += vals[e] * H[(long)cols[e] * D + t];   // L2-resident gather
    }
    if (cur >= 0) atomicAdd(&Out[(long)cur * D + t], acc);
}

// ============================================================
// log_softmax over dim 40, in place. One wave32 per row; 8 rows/block.
// ============================================================
__global__ void logsoftmax40(float* __restrict__ Out, int nRows) {
    const int lane = threadIdx.x & 31;
    const int row  = blockIdx.x * 8 + (threadIdx.x >> 5);
    if (row >= nRows) return;
    float* p = Out + (long)row * OUT_DIM;

    float v0 = p[lane];                                   // cols 0..31
    float v1 = (lane < 8) ? p[32 + lane] : -3.4e38f;      // cols 32..39

    float m = fmaxf(v0, v1);
#pragma unroll
    for (int off = 16; off; off >>= 1)
        m = fmaxf(m, __shfl_xor(m, off, 32));

    float s = __expf(v0 - m) + ((lane < 8) ? __expf(v1 - m) : 0.0f);
#pragma unroll
    for (int off = 16; off; off >>= 1)
        s += __shfl_xor(s, off, 32);

    float lse = m + __logf(s);
    p[lane] = v0 - lse;
    if (lane < 8) p[32 + lane] = v1 - lse;
}

// ============================================================
// launcher
// ============================================================
extern "C" void kernel_launch(void* const* d_in, const int* in_sizes, int n_in,
                              void* d_out, int out_size, void* d_ws, size_t ws_size,
                              hipStream_t stream) {
    const float* x     = (const float*)d_in[0];
    const float* W1    = (const float*)d_in[1];
    const float* W2    = (const float*)d_in[2];
    const float* evals = (const float*)d_in[3];
    const int*   erows = (const int*)d_in[4];
    const int*   ecols = (const int*)d_in[5];
    float* out = (float*)d_out;

    const int N = in_sizes[0] / IN_DIM;   // 100000
    const int E = in_sizes[3];            // 3200000

    float* ws = (float*)d_ws;
    float* h0 = ws;                               // [N,128]  51.2 MB
    float* h1 = ws + (size_t)N * HID_DIM;         // [N,128]  51.2 MB
    float* h2 = ws;                               // [N,40] — reuses h0 (dead by then)
    __bf16* bp1 = (__bf16*)(ws + (size_t)2 * N * HID_DIM);        // 128 KB, 32B-aligned
    __bf16* bp2 = bp1 + (size_t)NTILES1 * KSTEPS1 * 32 * 16;      // 12 KB

    const int CHUNK = 512;
    const int mTiles = (N + 15) / 16;

    // zero accumulation targets
    zero_f32<<<1024, 256, 0, stream>>>(h1, (long)N * HID_DIM);
    zero_f32<<<512, 256, 0, stream>>>(out, (long)N * OUT_DIM);

    // pack B fragments (zero-padded K tail / col tail)
    packB<<<(NTILES1 * KSTEPS1 * 32 + 255) / 256, 256, 0, stream>>>(
        W1, bp1, IN_DIM, HID_DIM, NTILES1, KSTEPS1);
    packB<<<(NTILES2 * KSTEPS2 * 32 + 255) / 256, 256, 0, stream>>>(
        W2, bp2, HID_DIM, OUT_DIM, NTILES2, KSTEPS2);

    // layer 1: dense GEMM (bf16 WMMA) then SpMM
    gemm1_wmma<<<dim3(mTiles, 2), 128, 0, stream>>>(x, bp1, h0, N);
    spmm_coo<HID_DIM><<<(E + CHUNK - 1) / CHUNK, 128, 0, stream>>>(
        erows, ecols, evals, h0, h1, E, CHUNK);

    // layer 2: relu fused into GEMM A-load, then SpMM into d_out
    gemm2_wmma<<<mTiles, 96, 0, stream>>>(h1, bp2, h2, N);
    spmm_coo<OUT_DIM><<<(E + CHUNK - 1) / CHUNK, 64, 0, stream>>>(
        erows, ecols, evals, h2, out, E, CHUNK);

    // in-place log_softmax over dim=40
    logsoftmax40<<<(N + 7) / 8, 256, 0, stream>>>(out, N);
}
